// _NonLocalBlockND_30193620091297
// MI455X (gfx1250) — matrix operand order repaired and checked by reading, not compiled, not run on hardware
//
#include <hip/hip_runtime.h>

// ---------------------------------------------------------------------------
// NonLocalBlockND (dual-branch non-local attention) for MI455X / gfx1250.
// B=4, C=64, CI=32, H=W=96. Attention computed at 48x48 (theta is up2() of a
// stride-2 conv => rows of f repeat 2x2; y/BN are replication-invariant).
// Attention matmuls use v_wmma_f32_16x16x32_f16 (K=32 == CI, one WMMA/tile).
// g is staged in a chunk-swizzled layout so PV B-fragments load contiguously.
// ---------------------------------------------------------------------------

typedef __attribute__((ext_vector_type(16))) _Float16 v16h;
typedef __attribute__((ext_vector_type(8)))  float    v8f;

#define BB   4
#define CC   64
#define CI   32
#define HH   96
#define WW   96
#define HL   48           // low-res grid
#define NL   (HL*HL)      // 2304
#define NT   (NL/16)      // 144 tiles of 16 rows
#define KC   (NL/32)      // 72 K=32 chunks for the PV matmul

// ---------------- Stage A: convolutions -> f16 staging buffers -------------

// 1x1 conv (C->CI) + 2x2 maxpool; store CHUNK-SWIZZLED gsw[b][m/32][ci][m%32]
// so the PV-matmul B-fragment is 16 contiguous f16 per lane.
__global__ __launch_bounds__(256) void k_g(const float* __restrict__ x,
                                           const float* __restrict__ w,
                                           const float* __restrict__ bias,
                                           _Float16* __restrict__ gsw) {
  int idx = blockIdx.x * 256 + threadIdx.x;
  if (idx >= BB * NL * CI) return;
  int ci = idx & 31;
  int m  = (idx >> 5) % NL;
  int b  = idx / (NL * CI);
  int py = m / HL, px = m % HL;
  const float* wp = w + ci * CC;
  float best = -3.0e38f;
  for (int dy = 0; dy < 2; ++dy)
    for (int dx = 0; dx < 2; ++dx) {
      int iy = py * 2 + dy, ix = px * 2 + dx;
      const float* xp = x + (((size_t)b * CC) * HH + iy) * WW + ix;
      float s = 0.f;
      for (int c = 0; c < CC; ++c) s += wp[c] * xp[(size_t)c * HH * WW];
      best = fmaxf(best, s);
    }
  // swizzled: [b][kc=m/32][ci][r=m%32]
  gsw[(((size_t)b * KC + (m >> 5)) * CI + ci) * 32 + (m & 31)] =
      (_Float16)(best + bias[ci]);
}

// 3x3 conv stride2 pad1 (C->CI) -> theta[b][n][ci] f16 (48x48 grid, pre-up2).
__global__ __launch_bounds__(256) void k_theta(const float* __restrict__ x,
                                               const float* __restrict__ w,
                                               const float* __restrict__ bias,
                                               _Float16* __restrict__ th) {
  int idx = blockIdx.x * 256 + threadIdx.x;
  if (idx >= BB * NL * CI) return;
  int ci = idx & 31;
  int n  = (idx >> 5) % NL;
  int b  = idx / (NL * CI);
  int oy = n / HL, ox = n % HL;
  float s = bias[ci];
  for (int ky = 0; ky < 3; ++ky) {
    int iy = oy * 2 - 1 + ky;
    if (iy < 0 || iy >= HH) continue;
    for (int kx = 0; kx < 3; ++kx) {
      int ix = ox * 2 - 1 + kx;
      if (ix < 0 || ix >= WW) continue;
      const float* wp = w + ((ci * CC) * 3 + ky) * 3 + kx;
      const float* xp = x + (((size_t)b * CC) * HH + iy) * WW + ix;
      for (int c = 0; c < CC; ++c)
        s += wp[(size_t)c * 9] * xp[(size_t)c * HH * WW];
    }
  }
  th[((size_t)b * NL + n) * CI + ci] = (_Float16)s;
}

// 3x3 conv stride2 pad1 + 2x2 maxpool (->24x24) + nearest up2 back to 48x48;
// store phi[b][m][ci] f16 (each pooled value replicated to 4 m positions).
__global__ __launch_bounds__(256) void k_phi(const float* __restrict__ x,
                                             const float* __restrict__ w,
                                             const float* __restrict__ bias,
                                             _Float16* __restrict__ ph) {
  int idx = blockIdx.x * 256 + threadIdx.x;
  if (idx >= BB * 576 * CI) return;
  int ci = idx & 31;
  int p  = (idx >> 5) % 576;   // 24x24 pooled grid
  int b  = idx / (576 * CI);
  int py = p / 24, px = p % 24;
  float best = -3.0e38f;
  for (int dy = 0; dy < 2; ++dy)
    for (int dx = 0; dx < 2; ++dx) {
      int oy = py * 2 + dy, ox = px * 2 + dx;   // 48x48 conv grid
      float s = bias[ci];
      for (int ky = 0; ky < 3; ++ky) {
        int iy = oy * 2 - 1 + ky;
        if (iy < 0 || iy >= HH) continue;
        for (int kx = 0; kx < 3; ++kx) {
          int ix = ox * 2 - 1 + kx;
          if (ix < 0 || ix >= WW) continue;
          const float* wp = w + ((ci * CC) * 3 + ky) * 3 + kx;
          const float* xp = x + (((size_t)b * CC) * HH + iy) * WW + ix;
          for (int c = 0; c < CC; ++c)
            s += wp[(size_t)c * 9] * xp[(size_t)c * HH * WW];
        }
      }
      best = fmaxf(best, s);
    }
  _Float16 v = (_Float16)best;
  for (int dy = 0; dy < 2; ++dy)
    for (int dx = 0; dx < 2; ++dx) {
      int m = (py * 2 + dy) * HL + (px * 2 + dx);
      ph[((size_t)b * NL + m) * CI + ci] = v;
    }
}

// ---------------- Stage B: fused attention (WMMA) --------------------------
// One workgroup per (batch, 16-row n-tile). 8 waves.
// LDS: sc[16][2304] f32 (147456B) | prob[16][2304] f16 (73728B) | red (1088B)
__global__ __launch_bounds__(256) void k_attn(const _Float16* __restrict__ th,
                                              const _Float16* __restrict__ ph,
                                              const _Float16* __restrict__ gsw,
                                              float* __restrict__ y) {
  extern __shared__ char smem[];
  float*    sc      = (float*)smem;                        // 16*2304 f32
  _Float16* prob    = (_Float16*)(smem + 16 * NL * 4);     // 16*2304 f16
  float*    red     = (float*)(smem + 16 * NL * 4 + 16 * NL * 2);  // 256 f32
  float*    rowstat = red + 256;                           // 16 f32

  int tid  = threadIdx.x;
  int wave = tid >> 5;
  int lane = tid & 31;
  int lrow = lane & 15;
  int hi   = lane >> 4;                 // half-wave select
  int b    = blockIdx.x / NT;
  int n0   = (blockIdx.x % NT) * 16;

  // --- A fragment: theta rows n0..n0+15, K=0..31 (contiguous K = 16*hi + e order, same as B) ---
  // lanes 0-15: elems 0-7 -> K 0-7, elems 8-15 -> K 8-15
  // lanes16-31: elems 0-7 -> K 16-23, elems 8-15 -> K 24-31
  int kb0 = hi ? 16 : 0, kb1 = hi ? 24 : 8;
  v16h afrag;
  {
    const _Float16* tp = th + ((size_t)b * NL + n0 + lrow) * CI;
    for (int e = 0; e < 8; ++e) { afrag[e] = tp[kb0 + e]; afrag[8 + e] = tp[kb1 + e]; }
  }

  // --- Phase 1: scores f[n0..n0+15][:] = theta_tile^T phi, one WMMA/tile ---
  for (int mt = wave; mt < NT; mt += 8) {
    int m0 = mt * 16;
    // B fragment (32x16): lane = column m0+lrow; K = hi*16 + e
    const _Float16* pp = ph + ((size_t)b * NL + m0 + lrow) * CI + (hi ? 16 : 0);
    v16h bfrag;
    for (int e = 0; e < 16; ++e) bfrag[e] = pp[e];
    v8f acc = {};
    acc = __builtin_amdgcn_wmma_f32_16x16x32_f16(false, afrag, false, bfrag,
                                                 (short)0, acc, false, false);
    // D layout: lane<16: VGPR r -> (row r, col lrow); lane>=16: (row r+8, col lrow)
    int colm = m0 + lrow;
    for (int r = 0; r < 8; ++r) sc[(r + hi * 8) * NL + colm] = acc[r];
  }
  __syncthreads();

  // --- Phase 2: softmax over m (16 threads per row) ---
  int srow = tid >> 4;
  int ssub = tid & 15;
  float mx = -3.0e38f;
  for (int m = ssub; m < NL; m += 16) mx = fmaxf(mx, sc[srow * NL + m]);
  red[srow * 16 + ssub] = mx;
  __syncthreads();
  if (ssub == 0) {
    float v = red[srow * 16];
    for (int i = 1; i < 16; ++i) v = fmaxf(v, red[srow * 16 + i]);
    rowstat[srow] = v;
  }
  __syncthreads();
  float rmax = rowstat[srow];
  float sm = 0.f;
  for (int m = ssub; m < NL; m += 16) {
    float e = __expf(sc[srow * NL + m] - rmax);
    sc[srow * NL + m] = e;
    sm += e;
  }
  red[srow * 16 + ssub] = sm;
  __syncthreads();
  if (ssub == 0) {
    float v = 0.f;
    for (int i = 0; i < 16; ++i) v += red[srow * 16 + i];
    rowstat[srow] = 1.0f / v;
  }
  __syncthreads();
  float rinv = rowstat[srow];
  for (int m = ssub; m < NL; m += 16)
    prob[srow * NL + m] = (_Float16)(sc[srow * NL + m] * rinv);
  __syncthreads();

  // --- Phase 3: y_tile(16x32) = prob(16x2304) @ g^T(2304x32), K=32 chunks ---
  // gsw layout [b][kc][ci][r]: lane reads 16 contiguous f16 per fragment.
  v8f y0 = {}, y1 = {};
  for (int kc = wave; kc < KC; kc += 8) {
    int mk = kc * 32;
    v16h pa;
    const _Float16* pp = prob + lrow * NL + mk;
    for (int e = 0; e < 8; ++e) { pa[e] = pp[kb0 + e]; pa[8 + e] = pp[kb1 + e]; }
    const _Float16* gp =
        gsw + (((size_t)b * KC + kc) * CI + lrow) * 32 + hi * 16;
    __builtin_prefetch(gp + 8 * CI * 32, 0, 1);   // next chunk for this wave
    v16h gb0, gb1;
    for (int e = 0; e < 16; ++e) { gb0[e] = gp[e]; gb1[e] = gp[16 * 32 + e]; }
    y0 = __builtin_amdgcn_wmma_f32_16x16x32_f16(false, pa, false, gb0,
                                                (short)0, y0, false, false);
    y1 = __builtin_amdgcn_wmma_f32_16x16x32_f16(false, pa, false, gb1,
                                                (short)0, y1, false, false);
  }
  __syncthreads();                       // done reading prob; reuse sc region

  // --- Phase 4: cross-wave reduction of partial y tiles ---
  float* yred = sc;                      // [8 waves][16 rows][32 cols]
  for (int r = 0; r < 8; ++r) {
    int row = r + hi * 8;
    yred[(wave * 16 + row) * 32 + lrow]      = y0[r];
    yred[(wave * 16 + row) * 32 + lrow + 16] = y1[r];
  }
  __syncthreads();
  for (int o = tid; o < 512; o += 256) {
    int row = o >> 5, c = o & 31;
    float s = 0.f;
    for (int wv = 0; wv < 8; ++wv) s += yred[(wv * 16 + row) * 32 + c];
    y[((size_t)b * NL + n0 + row) * CI + c] = s;
  }
}

// ---------------- Stage C: 1x1 conv W + deterministic BN --------------------

// z[b][cout][m] = sum_ci W[cout][ci]*y[b][m][ci] + b; per-block partial stats.
__global__ __launch_bounds__(256) void k_wconv(const float* __restrict__ y,
                                               const float* __restrict__ w,
                                               const float* __restrict__ bias,
                                               float* __restrict__ z,
                                               float* __restrict__ partial,
                                               int branch) {
  int blk  = blockIdx.x;                // B*64*9 blocks
  int mseg = blk % 9;
  int cout = (blk / 9) % CC;
  int b    = blk / (9 * CC);
  int t    = threadIdx.x;
  int m    = mseg * 256 + t;
  const float* wp = w + cout * CI;
  const float* yp = y + ((size_t)b * NL + m) * CI;
  float s = bias[cout];
  for (int ci = 0; ci < CI; ++ci) s += wp[ci] * yp[ci];
  z[((size_t)b * CC + cout) * NL + m] = s;

  __shared__ float r1[256], r2[256];
  r1[t] = s; r2[t] = s * s;
  __syncthreads();
  for (int st = 128; st > 0; st >>= 1) {
    if (t < st) { r1[t] += r1[t + st]; r2[t] += r2[t + st]; }
    __syncthreads();
  }
  if (t == 0) {
    int slot = b * 9 + mseg;            // 36 partials per channel
    partial[((branch * 2 + 0) * CC + cout) * 36 + slot] = r1[0];
    partial[((branch * 2 + 1) * CC + cout) * 36 + slot] = r2[0];
  }
}

// Reduce partials -> mean & invstd per (branch, channel). 128 threads.
__global__ void k_stats(const float* __restrict__ partial, float* __restrict__ mv) {
  int t = threadIdx.x;
  if (t >= 128) return;
  int br = t >> 6, c = t & 63;
  float s = 0.f, q = 0.f;
  for (int k = 0; k < 36; ++k) {
    s += partial[((br * 2 + 0) * CC + c) * 36 + k];
    q += partial[((br * 2 + 1) * CC + c) * 36 + k];
  }
  float cnt  = (float)(BB * NL);        // == full-res stats (2x2 replication)
  float mean = s / cnt;
  float var  = q / cnt - mean * mean;
  mv[(br * CC + c) * 2 + 0] = mean;
  mv[(br * CC + c) * 2 + 1] = rsqrtf(var + 1e-5f);
}

// Apply BN affine, nearest-up2 to 96x96, write concat output (B,128,96,96).
__global__ __launch_bounds__(256) void k_bn_out(const float* __restrict__ zS,
                                                const float* __restrict__ zP,
                                                const float* __restrict__ mv,
                                                const float* __restrict__ Wg,
                                                const float* __restrict__ Wbe,
                                                const float* __restrict__ Wpg,
                                                const float* __restrict__ Wpbe,
                                                float* __restrict__ out) {
  size_t idx = (size_t)blockIdx.x * 256 + threadIdx.x;
  if (idx >= (size_t)BB * 128 * HH * WW) return;
  int j  = idx % WW;
  size_t r = idx / WW;
  int i  = r % HH;  r /= HH;
  int ch = r % 128; int b = r / 128;
  int br = ch >> 6, c = ch & 63;
  const float* z = br ? zP : zS;
  float v    = z[((size_t)b * CC + c) * NL + (i >> 1) * HL + (j >> 1)];
  float mean = mv[(br * CC + c) * 2 + 0];
  float inv  = mv[(br * CC + c) * 2 + 1];
  float g    = br ? Wpg[c]  : Wg[c];
  float be   = br ? Wpbe[c] : Wbe[c];
  out[idx] = g * (v - mean) * inv + be;
}

// ---------------------------------------------------------------------------

extern "C" void kernel_launch(void* const* d_in, const int* in_sizes, int n_in,
                              void* d_out, int out_size, void* d_ws, size_t ws_size,
                              hipStream_t stream) {
  (void)in_sizes; (void)n_in; (void)out_size; (void)ws_size;
  const float* x     = (const float*)d_in[0];
  const float* x_pan = (const float*)d_in[1];
  const float* g_w   = (const float*)d_in[2];
  const float* g_b   = (const float*)d_in[3];
  const float* gp_w  = (const float*)d_in[4];
  const float* gp_b  = (const float*)d_in[5];
  const float* th_w  = (const float*)d_in[6];
  const float* th_b  = (const float*)d_in[7];
  const float* ph_w  = (const float*)d_in[8];
  const float* ph_b  = (const float*)d_in[9];
  const float* php_w = (const float*)d_in[10];
  const float* php_b = (const float*)d_in[11];
  const float* W_w   = (const float*)d_in[12];
  const float* W_b   = (const float*)d_in[13];
  const float* W_g   = (const float*)d_in[14];
  const float* W_be  = (const float*)d_in[15];
  const float* Wp_w  = (const float*)d_in[16];
  const float* Wp_b  = (const float*)d_in[17];
  const float* Wp_g  = (const float*)d_in[18];
  const float* Wp_be = (const float*)d_in[19];
  float* out = (float*)d_out;

  // workspace carve-up (~10 MB)
  char* ws = (char*)d_ws;
  float*    partial = (float*)ws;                       // 9216 f32
  float*    mv      = (float*)(ws + 36864);             // 256 f32
  _Float16* theta   = (_Float16*)(ws + 40960);
  const size_t NE   = (size_t)BB * NL * CI;             // 294912 elems
  _Float16* phiS = theta + NE;
  _Float16* phiP = phiS  + NE;
  _Float16* gS   = phiP  + NE;
  _Float16* gP   = gS    + NE;
  float*    yS   = (float*)(gP + NE);
  float*    yP   = yS + NE;
  float*    zS   = yP + NE;
  float*    zP   = zS + (size_t)BB * CC * NL;

  k_theta<<<1152, 256, 0, stream>>>(x,     th_w,  th_b,  theta);
  k_g    <<<1152, 256, 0, stream>>>(x,     g_w,   g_b,   gS);
  k_g    <<<1152, 256, 0, stream>>>(x_pan, gp_w,  gp_b,  gP);
  k_phi  <<< 288, 256, 0, stream>>>(x,     ph_w,  ph_b,  phiS);
  k_phi  <<< 288, 256, 0, stream>>>(x_pan, php_w, php_b, phiP);

  const size_t shmem = 16 * NL * 4 + 16 * NL * 2 + 256 * 4 + 16 * 4; // 222272 B
  k_attn<<<BB * NT, 256, shmem, stream>>>(theta, phiS, gS, yS);
  k_attn<<<BB * NT, 256, shmem, stream>>>(theta, phiP, gP, yP);

  k_wconv<<<BB * CC * 9, 256, 0, stream>>>(yS, W_w,  W_b,  zS, partial, 0);
  k_wconv<<<BB * CC * 9, 256, 0, stream>>>(yP, Wp_w, Wp_b, zP, partial, 1);
  k_stats<<<1, 128, 0, stream>>>(partial, mv);
  k_bn_out<<<18432, 256, 0, stream>>>(zS, zP, mv, W_g, W_be, Wp_g, Wp_be, out);
}
